// HierarchicalAttentionRNN_10325101379852
// MI455X (gfx1250) — matrix operand order repaired and checked by reading, not compiled, not run on hardware
//
#include <hip/hip_runtime.h>
#include <hip/hip_bf16.h>
#include <math.h>

typedef __attribute__((ext_vector_type(16))) _Float16 v16h;
typedef __attribute__((ext_vector_type(8)))  float    v8f;

#define DEV __device__ __forceinline__

// ---------------------------------------------------------------------------
// WMMA helpers: D = A(16x32 f16) * B(32x16 f16) + C(16x16 f32), wave32.
// ---------------------------------------------------------------------------
DEV v8f wmma16x16x32(v16h a, v16h b, v8f c) {
  return __builtin_amdgcn_wmma_f32_16x16x32_f16(
      /*neg_a=*/false, a, /*neg_b=*/false, b,
      /*c_mod=*/(short)0, c, /*reuse_a=*/false, /*reuse_b=*/false);
}

// A fragment from a row-major [16][ld] f16 tile (LDS or global).
// Per ISA 7.12.2: lanes 0-15 row M=lane, K groups {0..7,16..23};
// lanes 16-31 same rows, K groups {8..15,24..31}.
DEV v16h load_a_frag(const _Float16* base, int ld, int k0) {
  int lane = threadIdx.x & 31;
  int m = lane & 15;
  int kb = (lane < 16) ? 0 : 8;
  const _Float16* row = base + m * ld + k0;
  v16h a;
#pragma unroll
  for (int j = 0; j < 4; ++j) {
    a[2 * j]     = row[kb + 2 * j];
    a[2 * j + 1] = row[kb + 2 * j + 1];
    a[8 + 2 * j]     = row[16 + kb + 2 * j];
    a[8 + 2 * j + 1] = row[16 + kb + 2 * j + 1];
  }
  return a;
}

// B fragment from pre-swizzled weights: tile (nt,kc) is 512 contiguous f16,
// lane's 16 values contiguous (one 32B load).
DEV v16h load_b_frag(const _Float16* sw, int KC, int nt, int kc) {
  int lane = threadIdx.x & 31;
  const _Float16* p = sw + (((size_t)(nt * KC + kc)) << 9) + lane * 16;
  return *(const v16h*)p;
}

// C/D layout (ISA 7.12.2): lane L -> N=L%16, VGPR j -> M=j+8*(L/16).
DEV void store_acc(float* dst, int ld, int n0, v8f c) {
  int lane = threadIdx.x & 31;
  int n = n0 + (lane & 15);
  int mh = (lane >> 4) << 3;
#pragma unroll
  for (int j = 0; j < 8; ++j) dst[(mh + j) * ld + n] = c[j];
}

DEV float sigmoidf_(float x) { return 1.0f / (1.0f + __expf(-x)); }

// ---------------------------------------------------------------------------
// Weight swizzle: W row-major [N][K] f32  ->  B-fragment-ordered f16.
// dst element idx encodes (tile = nt*KC+kc, lane, j, p); value = W[n][k]
// with n = nt*16 + lane%16, k = kc*32 + (lane<16?0:8) + (r<8 ? r : r+8), r = 2j + p.
// ---------------------------------------------------------------------------
__global__ void swizzle_b_kernel(const float* __restrict__ W,
                                 _Float16* __restrict__ dst, int Nn, int K) {
  int KC = K >> 5;
  int total = Nn * K;
  for (int idx = blockIdx.x * blockDim.x + threadIdx.x; idx < total;
       idx += gridDim.x * blockDim.x) {
    int tile = idx >> 9, within = idx & 511;
    int lane = within >> 4, r = within & 15;
    int j = r >> 1, p = r & 1;
    int kc = tile % KC, nt = tile / KC;
    int n = nt * 16 + (lane & 15);
    int k = kc * 32 + ((lane < 16) ? 0 : 8) + ((r < 8) ? r : r + 8);
    dst[idx] = (_Float16)W[(size_t)n * K + k];
  }
}

// Embedding gather, f32 -> f16: xh[nt*64+e] = emb[chars[nt]*64+e]
__global__ void embed_kernel(const int* __restrict__ chars,
                             const float* __restrict__ emb,
                             _Float16* __restrict__ xh, int total) {
  for (int idx = blockIdx.x * blockDim.x + threadIdx.x; idx < total;
       idx += gridDim.x * blockDim.x) {
    int nt = idx >> 6, e = idx & 63;
    xh[idx] = (_Float16)emb[(size_t)chars[nt] * 64 + e];
  }
}

// ---------------------------------------------------------------------------
// Masked (optionally reversed-within-length) GRU direction.
// One workgroup (8 waves) owns 16 sequences; h kept in LDS across T steps.
// Waves 0-3: fused r/z gates GEMM over K = DIN+128 (x and h parts add).
// Waves 4-7: n-gate with SEPARATE x / h accumulators (needed for xn + r*hn).
// Output layout: out[n][t][256], this direction writes [outOff .. outOff+128).
// ---------------------------------------------------------------------------
template <int DIN>
__global__ __launch_bounds__(256) void gru_dir_kernel(
    const _Float16* __restrict__ x,      // [N][T][DIN] f16
    const int* __restrict__ lens,        // [N] or nullptr (=> len = T)
    const _Float16* __restrict__ wihSw,  // swizzled, N=384, K=DIN
    const _Float16* __restrict__ whhSw,  // swizzled, N=384, K=128
    const float* __restrict__ bih, const float* __restrict__ bhh,
    _Float16* __restrict__ out,          // [N][T][256]
    int T, int reverse, int outOff) {
  constexpr int KCX = DIN / 32;
  __shared__ _Float16 sX[16 * DIN];
  __shared__ _Float16 sH[16 * 128];
  __shared__ float sRZ[16 * 256];
  __shared__ float sXN[16 * 128];
  __shared__ float sHN[16 * 128];
  __shared__ float sBih[384], sBhh[384];
  __shared__ int sLen[16], sSrc[16];

  const int tid = threadIdx.x;
  const int n0 = blockIdx.x * 16;
  for (int i = tid; i < 384; i += 256) { sBih[i] = bih[i]; sBhh[i] = bhh[i]; }
  for (int i = tid; i < 16 * 128; i += 256) sH[i] = (_Float16)0.0f;
  if (tid < 16) sLen[tid] = lens ? lens[n0 + tid] : T;
  __syncthreads();

  const int wv = tid >> 5;

  for (int s = 0; s < T; ++s) {
    if (tid < 16) {
      int L = sLen[tid];
      sSrc[tid] = (s < L) ? (reverse ? (L - 1 - s) : s) : s;
    }
    __syncthreads();
    for (int i = tid; i < 16 * DIN; i += 256) {
      int m = i / DIN, e = i - m * DIN;
      sX[i] = x[((size_t)(n0 + m) * T + sSrc[m]) * DIN + e];
    }
    __syncthreads();

    if (wv < 4) {  // r,z gates: column tiles 0..15, 4 per wave, fused K.
      v8f acc[4] = {};
#pragma unroll
      for (int kc = 0; kc < KCX; ++kc) {
        v16h a = load_a_frag(sX, DIN, kc * 32);
#pragma unroll
        for (int i = 0; i < 4; ++i)
          acc[i] = wmma16x16x32(a, load_b_frag(wihSw, KCX, wv * 4 + i, kc), acc[i]);
      }
#pragma unroll
      for (int kc = 0; kc < 4; ++kc) {
        v16h a = load_a_frag(sH, 128, kc * 32);
#pragma unroll
        for (int i = 0; i < 4; ++i)
          acc[i] = wmma16x16x32(a, load_b_frag(whhSw, 4, wv * 4 + i, kc), acc[i]);
      }
#pragma unroll
      for (int i = 0; i < 4; ++i) store_acc(sRZ, 256, (wv * 4 + i) * 16, acc[i]);
    } else {  // n gate: column tiles 16..23, 2 per wave, split x/h accums.
      const int w = wv - 4;
      v8f ax[2] = {}, ah[2] = {};
#pragma unroll
      for (int kc = 0; kc < KCX; ++kc) {
        v16h a = load_a_frag(sX, DIN, kc * 32);
#pragma unroll
        for (int i = 0; i < 2; ++i)
          ax[i] = wmma16x16x32(a, load_b_frag(wihSw, KCX, 16 + w * 2 + i, kc), ax[i]);
      }
#pragma unroll
      for (int kc = 0; kc < 4; ++kc) {
        v16h a = load_a_frag(sH, 128, kc * 32);
#pragma unroll
        for (int i = 0; i < 2; ++i)
          ah[i] = wmma16x16x32(a, load_b_frag(whhSw, 4, 16 + w * 2 + i, kc), ah[i]);
      }
#pragma unroll
      for (int i = 0; i < 2; ++i) {
        store_acc(sXN, 128, (w * 2 + i) * 16, ax[i]);
        store_acc(sHN, 128, (w * 2 + i) * 16, ah[i]);
      }
    }
    __syncthreads();

    // Gate nonlinearities + state update (f32 VALU), 8 elements/thread.
    for (int i = tid; i < 16 * 128; i += 256) {
      int m = i >> 7, j = i & 127;
      float r = sigmoidf_(sRZ[m * 256 + j] + sBih[j] + sBhh[j]);
      float z = sigmoidf_(sRZ[m * 256 + 128 + j] + sBih[128 + j] + sBhh[128 + j]);
      float nv = tanhf(sXN[i] + sBih[256 + j] + r * (sHN[i] + sBhh[256 + j]));
      float hOld = (float)sH[i];
      float hNew = (1.0f - z) * nv + z * hOld;
      bool valid = s < sLen[m];
      // valid step: write at src (fwd: s, bwd: L-1-s). invalid: zero at s.
      out[((size_t)(n0 + m) * T + sSrc[m]) * 256 + outOff + j] =
          (_Float16)(valid ? hNew : 0.0f);
      sH[i] = (_Float16)(valid ? hNew : hOld);
    }
    __syncthreads();
  }
}

// ---------------------------------------------------------------------------
// Attention pool over one sequence: proj = tanh(out @ Wp^T + bp) via WMMA
// (M=T, N=128, K=256), s[t] = proj[t].ctx, softmax over all T, weighted sum.
// ---------------------------------------------------------------------------
template <int T>
__global__ __launch_bounds__(256) void attn_pool_kernel(
    const _Float16* __restrict__ seqAll,  // [N][T][256]
    const _Float16* __restrict__ wpSw,    // swizzled, N=128, K=256
    const float* __restrict__ bp, const float* __restrict__ ctx,
    float* __restrict__ outF32, _Float16* __restrict__ outF16) {
  constexpr int TM = T / 16;
  __shared__ float sP[T * 128];
  __shared__ float sS[T];
  __shared__ float sAtt[T];
  const int tid = threadIdx.x;
  const int wv = tid >> 5;
  const int lane = tid & 31;
  const _Float16* seq = seqAll + (size_t)blockIdx.x * T * 256;

#pragma unroll
  for (int i = 0; i < TM; ++i) {
    int tile = wv * TM + i;  // TM*8 output tiles total, TM per wave
    int mt = tile >> 3, nt = tile & 7;
    v8f acc = {};
#pragma unroll
    for (int kc = 0; kc < 8; ++kc)
      acc = wmma16x16x32(load_a_frag(seq + (size_t)mt * 16 * 256, 256, kc * 32),
                         load_b_frag(wpSw, 8, nt, kc), acc);
    int n = nt * 16 + (lane & 15);
    int mh = (lane >> 4) << 3;
    float cv = ctx[n], bv = bp[n];
#pragma unroll
    for (int j = 0; j < 8; ++j)
      sP[(mt * 16 + mh + j) * 128 + n] = tanhf(acc[j] + bv) * cv;
  }
  __syncthreads();
  if (tid < T) {  // deterministic reduction over the 128 context dims
    float sum = 0.0f;
    for (int n = 0; n < 128; ++n) sum += sP[tid * 128 + n];
    sS[tid] = sum;
  }
  __syncthreads();
  if (tid == 0) {  // softmax over all T (pads contribute out=0, faithful)
    float mx = sS[0];
    for (int t = 1; t < T; ++t) mx = fmaxf(mx, sS[t]);
    float den = 0.0f;
    for (int t = 0; t < T; ++t) { float e = __expf(sS[t] - mx); sAtt[t] = e; den += e; }
    float inv = 1.0f / den;
    for (int t = 0; t < T; ++t) sAtt[t] *= inv;
  }
  __syncthreads();
  float acc = 0.0f;
  for (int t = 0; t < T; ++t) acc += sAtt[t] * (float)seq[t * 256 + tid];
  if (outF32) outF32[(size_t)blockIdx.x * 256 + tid] = acc;
  if (outF16) outF16[(size_t)blockIdx.x * 256 + tid] = (_Float16)acc;
}

// outputs = messages @ Wout^T + bout   ([128,256]x[256,8], tiny -> VALU)
__global__ void out_linear_kernel(const float* __restrict__ msgs,
                                  const float* __restrict__ Wout,
                                  const float* __restrict__ bout,
                                  float* __restrict__ out) {
  int i = blockIdx.x * blockDim.x + threadIdx.x;
  if (i >= 128 * 8) return;
  int b = i >> 3, o = i & 7;
  float a = bout[o];
  for (int k = 0; k < 256; ++k) a += msgs[b * 256 + k] * Wout[o * 256 + k];
  out[i] = a;
}

// ---------------------------------------------------------------------------
extern "C" void kernel_launch(void* const* d_in, const int* in_sizes, int n_in,
                              void* d_out, int out_size, void* d_ws,
                              size_t ws_size, hipStream_t stream) {
  (void)in_sizes; (void)n_in; (void)out_size; (void)ws_size;
  const int*   chars  = (const int*)d_in[0];
  const int*   lens   = (const int*)d_in[1];
  const float* emb    = (const float*)d_in[2];
  const float* cWih_f = (const float*)d_in[3];
  const float* cWhh_f = (const float*)d_in[4];
  const float* cbih_f = (const float*)d_in[5];
  const float* cbhh_f = (const float*)d_in[6];
  const float* cWih_b = (const float*)d_in[7];
  const float* cWhh_b = (const float*)d_in[8];
  const float* cbih_b = (const float*)d_in[9];
  const float* cbhh_b = (const float*)d_in[10];
  const float* mWih_f = (const float*)d_in[11];
  const float* mWhh_f = (const float*)d_in[12];
  const float* mbih_f = (const float*)d_in[13];
  const float* mbhh_f = (const float*)d_in[14];
  const float* mWih_b = (const float*)d_in[15];
  const float* mWhh_b = (const float*)d_in[16];
  const float* mbih_b = (const float*)d_in[17];
  const float* mbhh_b = (const float*)d_in[18];
  const float* wWp    = (const float*)d_in[19];
  const float* wbp    = (const float*)d_in[20];
  const float* wctx   = (const float*)d_in[21];
  const float* pWp    = (const float*)d_in[22];
  const float* pbp    = (const float*)d_in[23];
  const float* pctx   = (const float*)d_in[24];
  const float* Wout   = (const float*)d_in[25];
  const float* bout   = (const float*)d_in[26];

  char* ws = (char*)d_ws;
  size_t off = 0;
  auto alloc = [&](size_t bytes) -> void* {
    void* p = ws + off;
    off += (bytes + 255) & ~(size_t)255;
    return p;
  };
  const int NSEQ = 128 * 64;  // 8192 words
  _Float16* xh       = (_Float16*)alloc((size_t)NSEQ * 48 * 64 * 2);
  _Float16* charOut  = (_Float16*)alloc((size_t)NSEQ * 48 * 256 * 2);
  _Float16* words    = (_Float16*)alloc((size_t)NSEQ * 256 * 2);
  _Float16* msgOut   = (_Float16*)alloc((size_t)128 * 64 * 256 * 2);
  _Float16* cWihSw_f = (_Float16*)alloc(384 * 64 * 2);
  _Float16* cWhhSw_f = (_Float16*)alloc(384 * 128 * 2);
  _Float16* cWihSw_b = (_Float16*)alloc(384 * 64 * 2);
  _Float16* cWhhSw_b = (_Float16*)alloc(384 * 128 * 2);
  _Float16* mWihSw_f = (_Float16*)alloc(384 * 256 * 2);
  _Float16* mWhhSw_f = (_Float16*)alloc(384 * 128 * 2);
  _Float16* mWihSw_b = (_Float16*)alloc(384 * 256 * 2);
  _Float16* mWhhSw_b = (_Float16*)alloc(384 * 128 * 2);
  _Float16* wWpSw    = (_Float16*)alloc(128 * 256 * 2);
  _Float16* pWpSw    = (_Float16*)alloc(128 * 256 * 2);

  auto sw = [&](const float* src, _Float16* dst, int N, int K) {
    int total = N * K;
    swizzle_b_kernel<<<(total + 255) / 256, 256, 0, stream>>>(src, dst, N, K);
  };
  sw(cWih_f, cWihSw_f, 384, 64);
  sw(cWhh_f, cWhhSw_f, 384, 128);
  sw(cWih_b, cWihSw_b, 384, 64);
  sw(cWhh_b, cWhhSw_b, 384, 128);
  sw(mWih_f, mWihSw_f, 384, 256);
  sw(mWhh_f, mWhhSw_f, 384, 128);
  sw(mWih_b, mWihSw_b, 384, 256);
  sw(mWhh_b, mWhhSw_b, 384, 128);
  sw(wWp, wWpSw, 128, 256);
  sw(pWp, pWpSw, 128, 256);

  int embTotal = NSEQ * 48 * 64;
  embed_kernel<<<(embTotal + 255) / 256, 256, 0, stream>>>(chars, emb, xh, embTotal);

  // char BiGRU: 512 tiles of 16 sequences
  gru_dir_kernel<64><<<NSEQ / 16, 256, 0, stream>>>(
      xh, lens, cWihSw_f, cWhhSw_f, cbih_f, cbhh_f, charOut, 48, 0, 0);
  gru_dir_kernel<64><<<NSEQ / 16, 256, 0, stream>>>(
      xh, lens, cWihSw_b, cWhhSw_b, cbih_b, cbhh_b, charOut, 48, 1, 128);

  // word attention pool -> words (f16 input for message GRU)
  attn_pool_kernel<48><<<NSEQ, 256, 0, stream>>>(charOut, wWpSw, wbp, wctx,
                                                 nullptr, words);

  // message BiGRU (full length T=64, lens=nullptr)
  gru_dir_kernel<256><<<128 / 16, 256, 0, stream>>>(
      words, nullptr, mWihSw_f, mWhhSw_f, mbih_f, mbhh_f, msgOut, 64, 0, 0);
  gru_dir_kernel<256><<<128 / 16, 256, 0, stream>>>(
      words, nullptr, mWihSw_b, mWhhSw_b, mbih_b, mbhh_b, msgOut, 64, 1, 128);

  // message attention pool -> messages (d_out tuple part 2, f32)
  float* messages = (float*)d_out + 128 * 8;
  attn_pool_kernel<64><<<128, 256, 0, stream>>>(msgOut, pWpSw, pbp, pctx,
                                                messages, nullptr);

  // outputs (d_out tuple part 1)
  out_linear_kernel<<<4, 256, 0, stream>>>(messages, Wout, bout, (float*)d_out);
}
